// EchoStateNetwork_65627100283216
// MI455X (gfx1250) — matrix operand order repaired and checked by reading, not compiled, not run on hardware
//
#include <hip/hip_runtime.h>
#include <hip/hip_bf16.h>
#include <hip/hip_fp16.h>

typedef __attribute__((ext_vector_type(16))) _Float16 v16h;
typedef __attribute__((ext_vector_type(8)))  _Float16 v8h;
typedef __attribute__((ext_vector_type(8)))  float    v8f;

#define ESN_BATCH   256
#define ESN_T       512
#define ESN_C       6
#define ESN_R       256
#define BTILE       16                 // batch rows per workgroup
#define HPITCH      264                // h row pitch in halves (pad vs 256)
#define XROW        (ESN_T * ESN_C)    // 3072 floats per batch row

// Branch-free tanh. gfx1250 has a native TRANS32 tanh op; fall back to a
// clamp + v_exp_f32 + v_rcp_f32 identity (no EXEC-divergent libm path).
__device__ __forceinline__ float fast_tanh(float x) {
#if __has_builtin(__builtin_amdgcn_tanhf)
    return __builtin_amdgcn_tanhf(x);
#else
    float xc = fminf(fmaxf(x, -10.0f), 10.0f);       // tanh saturates ~1e-8 by |x|=10
    float e  = __expf(2.0f * xc);                    // v_exp_f32 based, finite
    return (e - 1.0f) * __builtin_amdgcn_rcpf(e + 1.0f);
#endif
}

// One workgroup = 16 batch rows, 8 waves. Wave w owns reservoir columns
// [32w, 32w+32) as two 16-wide N-tiles. W_res lives in registers as f16 WMMA
// B-fragments (128 VGPRs/wave). h lives in LDS (f16, double buffered). x tile
// lives entirely in LDS (196 KB of the 320 KB WGP LDS).
__global__ __launch_bounds__(256, 1)
void esn_fused_kernel(const float* __restrict__ x,
                      const float* __restrict__ W_in,
                      const float* __restrict__ W_res,
                      float* __restrict__ out)
{
    __shared__ float    x_lds[BTILE * XROW];           // 196608 B
    __shared__ _Float16 h_buf[2][BTILE * HPITCH];      //  16896 B

    const int tid   = threadIdx.x;
    const int wv    = tid >> 5;
    const int lane  = tid & 31;
    const int half  = lane >> 4;       // 0: lanes 0-15, 1: lanes 16-31
    const int l16   = lane & 15;
    const int bbase = blockIdx.x * BTILE;

    // ---- stage x tile into LDS (coalesced, one-time) ----
    const float* xsrc = x + (size_t)bbase * XROW;
    for (int i = tid; i < BTILE * XROW; i += 256)
        x_lds[i] = xsrc[i];

    // ---- h0 = 0 ----
    for (int i = tid; i < BTILE * HPITCH; i += 256)
        h_buf[0][i] = (_Float16)0.0f;

    // ---- per-lane W_in columns for the two owned N-tiles ----
    const int n0 = 32 * wv + l16;      // global reservoir column, tile 0
    const int n1 = n0 + 16;            // tile 1
    float win0[ESN_C], win1[ESN_C];
    #pragma unroll
    for (int c = 0; c < ESN_C; ++c) {
        win0[c] = W_in[c * ESN_R + n0];
        win1[c] = W_in[c * ESN_R + n1];
    }

    // ---- W_res -> register-resident f16 B-fragments ----
    // ISA B layout (16-bit, 32x16): halves j=0..7 hold K = kb*32 + (half?8:0) + j,
    // halves j=8..15 hold K = kb*32 + 16 + (half?8:0) + (j-8), column N = lane&15.
    v16h bfrag0[8], bfrag1[8];
    const int kboff = half * 8;
    #pragma unroll
    for (int kb = 0; kb < 8; ++kb) {
        #pragma unroll
        for (int j = 0; j < 16; ++j) {
            const int k = kb * 32 + kboff + j + (j >= 8 ? 8 : 0);
            bfrag0[kb][j] = (_Float16)W_res[k * ESN_R + n0];
            bfrag1[kb][j] = (_Float16)W_res[k * ESN_R + n1];
        }
    }

    __syncthreads();

    // ISA A layout (16-bit, 16x32): row M = lane&15; halves 0..7 = K+akoff+0..7,
    // halves 8..15 = K+16+akoff+0..7, akoff = (lane<16)?0:8.
    const int arow  = l16;
    const int akoff = half * 8;
    // ISA C/D layout (f32 16x16): VGPR v -> row v + (lane<16?0:8), col = lane&15.
    const int moff  = half * 8;

    int cur = 0;
    for (int t = 0; t < ESN_T; ++t) {
        // ---- prefetch ALL A-fragments into distinct registers so the 16
        //      ds_load_b128 pipeline behind a single dscnt wait, instead of
        //      load->wait->wmma->load->wait serialization ----
        const _Float16* hrow = &h_buf[cur][arow * HPITCH];
        v16h afrag[8];
        #pragma unroll
        for (int kb = 0; kb < 8; ++kb) {
            const v8h lo = *(const v8h*)(hrow + kb * 32 + akoff);
            const v8h hi = *(const v8h*)(hrow + kb * 32 + 16 + akoff);
            #pragma unroll
            for (int j = 0; j < 8; ++j) { afrag[kb][j] = lo[j]; afrag[kb][8 + j] = hi[j]; }
        }

        // ---- input projection xp_t directly into the C accumulators
        //      (VALU work that overlaps the A-fragment LDS latency) ----
        v8f acc0, acc1;
        #pragma unroll
        for (int v = 0; v < 8; ++v) {
            const float* xr = &x_lds[(v + moff) * XROW + t * ESN_C];  // LDS broadcast
            float s0 = 0.0f, s1 = 0.0f;
            #pragma unroll
            for (int c = 0; c < ESN_C; ++c) {
                const float xv = xr[c];
                s0 = fmaf(xv, win0[c], s0);
                s1 = fmaf(xv, win1[c], s1);
            }
            acc0[v] = s0;
            acc1[v] = s1;
        }

        // ---- h_t @ W_res : 16 WMMA back-to-back, two interleaved acc chains ----
        #pragma unroll
        for (int kb = 0; kb < 8; ++kb) {
            acc0 = __builtin_amdgcn_wmma_f32_16x16x32_f16(
                       false, afrag[kb], false, bfrag0[kb], (short)0, acc0, false, false);
            acc1 = __builtin_amdgcn_wmma_f32_16x16x32_f16(
                       false, afrag[kb], false, bfrag1[kb], (short)0, acc1, false, false);
        }

        // ---- tanh (branch-free), stream to global (NT), stash f16 h_{t+1} ----
        const int nxt = cur ^ 1;
        #pragma unroll
        for (int v = 0; v < 8; ++v) {
            const float h0v = fast_tanh(acc0[v]);
            const float h1v = fast_tanh(acc1[v]);
            const int m = v + moff;
            const size_t o = ((size_t)(bbase + m) * ESN_T + t) * ESN_R;
            __builtin_nontemporal_store(h0v, out + o + n0);
            __builtin_nontemporal_store(h1v, out + o + n1);
            h_buf[nxt][m * HPITCH + n0] = (_Float16)h0v;
            h_buf[nxt][m * HPITCH + n1] = (_Float16)h1v;
        }
        cur = nxt;
        __syncthreads();
    }
}

extern "C" void kernel_launch(void* const* d_in, const int* in_sizes, int n_in,
                              void* d_out, int out_size, void* d_ws, size_t ws_size,
                              hipStream_t stream) {
    (void)in_sizes; (void)n_in; (void)out_size; (void)d_ws; (void)ws_size;
    const float* x     = (const float*)d_in[0];
    const float* W_in  = (const float*)d_in[1];
    const float* W_res = (const float*)d_in[2];
    float* out = (float*)d_out;

    dim3 grid(ESN_BATCH / BTILE);   // 16 workgroups, one 16-row batch tile each
    dim3 block(256);                // 8 wave32
    hipLaunchKernelGGL(esn_fused_kernel, grid, block, 0, stream,
                       x, W_in, W_res, out);
}
